// WindowedSelfAttention3D_36859409334929
// MI455X (gfx1250) — matrix operand order repaired and checked by reading, not compiled, not run on hardware
//
#include <hip/hip_runtime.h>
#include <hip/hip_bf16.h>

typedef _Float16 half_t;
typedef __attribute__((ext_vector_type(16))) _Float16 v16h;
typedef __attribute__((ext_vector_type(8)))  _Float16 v8h;
typedef __attribute__((ext_vector_type(8)))  float    v8f;
typedef __attribute__((ext_vector_type(4)))  unsigned int u32x4;
typedef __attribute__((ext_vector_type(8)))  int          i32x8;
typedef __attribute__((ext_vector_type(4)))  int          i32x4;

#define C_CH  128
#define T_TOK 64
#define DHW   (64*64*64)
#define HW    (64*64)

// ---------- CDNA5 async global->LDS (per-lane scatter, tracked by ASYNCcnt) ----------
__device__ __forceinline__ void async_ld_b32(void* lds, const void* gptr) {
  unsigned int l = (unsigned int)(uintptr_t)lds;   // flat->LDS: addr[31:0]
  asm volatile("global_load_async_to_lds_b32 %0, %1, off"
               :: "v"(l), "v"(gptr) : "memory");
}
__device__ __forceinline__ void wait_async0() {
  asm volatile("s_wait_asynccnt 0x0" ::: "memory");
}

// ---------- WMMA fragment helpers (contiguous k order k = 16*(lane/16) + i, same for A and B) ----------

// A-matrix 16x32 f16 tile from row-major [rows][ld] f16.
// lane<16: row=m0+lane, K=k0+0..15 ; lane>=16: row=m0+lane-16, K=k0+16..31
__device__ __forceinline__ v16h load_A_tile(const half_t* src, int m0, int k0, int ld) {
  int lane = threadIdx.x & 31;
  int row  = m0 + (lane & 15);
  int kb   = k0 + ((lane & 16) ? 16 : 0);
  const half_t* p = src + row * ld + kb;
  v8h lo = *(const v8h*)(p);
  v8h hi = *(const v8h*)(p + 8);
  v16h a;
#pragma unroll
  for (int i = 0; i < 8; ++i) { a[i] = lo[i]; a[i + 8] = hi[i]; }
  return a;
}

// B-matrix 32x16 f16 tile where source is stored "transposed": src[n][k]
// (each B column is a contiguous K-run). lane<16: K=k0..k0+15; lane>=16: K=k0+16..k0+31.
__device__ __forceinline__ v16h load_B_tile_T(const half_t* src, int n0, int k0, int ld) {
  int lane = threadIdx.x & 31;
  int col  = n0 + (lane & 15);
  int kb   = k0 + ((lane & 16) ? 16 : 0);
  const half_t* p = src + col * ld + kb;
  v8h lo = *(const v8h*)(p);
  v8h hi = *(const v8h*)(p + 8);
  v16h b;
#pragma unroll
  for (int i = 0; i < 8; ++i) { b[i] = lo[i]; b[i + 8] = hi[i]; }
  return b;
}

// B-matrix 32x16 f16 tile from a K-major source: src[k][n] with row stride ld.
__device__ __forceinline__ v16h load_B_tile_KM(const half_t* src, int k0, int n0, int ld) {
  int lane = threadIdx.x & 31;
  int col  = n0 + (lane & 15);
  int kb   = k0 + ((lane & 16) ? 16 : 0);
  v16h b;
#pragma unroll
  for (int i = 0; i < 16; ++i) b[i] = src[(kb + i) * ld + col];
  return b;
}

__device__ __forceinline__ v8f wmma16(v16h a, v16h b, v8f c) {
  return __builtin_amdgcn_wmma_f32_16x16x32_f16(false, a, false, b, (short)0, c, false, false);
}

// ---------- kernel 0: weight fp32 -> f16 ----------
__global__ void __launch_bounds__(256)
convert_weights_kernel(const float* __restrict__ w_in, const float* __restrict__ w_out,
                       const float* __restrict__ w_conv, half_t* __restrict__ dst) {
  int i = blockIdx.x * 256 + threadIdx.x;
  if (i < 49152)        dst[i] = (half_t)w_in[i];
  else if (i < 65536)   dst[i] = (half_t)w_out[i - 49152];
  else if (i < 81920)   dst[i] = (half_t)w_conv[i - 65536];
}

// ---------- kernel 1: per-window LN + QKV + attention + out_proj + residual ----------
// y is produced in [B][spatial][C] f16 layout (channel-contiguous) so kernel 2 can
// fetch it with a single 2D TDM tile per block.
__global__ void __launch_bounds__(256)
win_attn_kernel(const float* __restrict__ x,
                const float* __restrict__ ln_g, const float* __restrict__ ln_b,
                const half_t* __restrict__ w_in, const float* __restrict__ b_in,
                const half_t* __restrict__ w_out, const float* __restrict__ b_out,
                half_t* __restrict__ y_h) {
  extern __shared__ char smem[];
  float*  xraw = (float*)(smem);            // [64][128] f32 : 32KB (pre-LN window, residual)
  half_t* xn   = (half_t*)(smem + 32768);   // [64][128] f16 : 16KB (post-LN)
  half_t* qkv  = (half_t*)(smem + 49152);   // [64][384] f16 : 48KB
  half_t* attn = (half_t*)(smem + 98304);   // [4][64][64] f16 : 32KB
  half_t* osh  = (half_t*)(smem + 131072);  // [64][128] f16 : 16KB (attention output)

  const int tid  = threadIdx.x;
  const int lane = tid & 31;
  const int wv   = tid >> 5;          // 8 waves
  const int hl   = (lane >> 4) & 1;
  const int ln16 = lane & 15;

  const int bid = blockIdx.x;         // 8192 windows
  const int b   = bid >> 12;
  const int wd  = (bid >> 8) & 15;
  const int wh  = (bid >> 4) & 15;
  const int ww  = bid & 15;
  const size_t xbase = (size_t)b * C_CH * DHW
                     + (size_t)(wd * 4) * HW + (wh * 4) * 64 + (ww * 4);

  if (tid == 0) __builtin_prefetch(w_in, 0, 0);   // global_prefetch_b8 (weights hot in L2)

  // ---- phase A: async-stage window -> xraw[t][c] (per-lane transposed scatter into LDS) ----
  for (int i = tid; i < T_TOK * C_CH; i += 256) {
    int c = i >> 6;
    int s = i & 63;
    int ld2 = s >> 4, lh = (s >> 2) & 3, lw = s & 3;
    async_ld_b32(&xraw[s * C_CH + c],
                 &x[xbase + (size_t)c * DHW + ld2 * HW + lh * 64 + lw]);
  }
  wait_async0();
  __syncthreads();

  // ---- phase B: LayerNorm over C, 8 tokens per wave, wave32 shuffle reduction ----
  for (int t8 = 0; t8 < 8; ++t8) {
    int t = wv * 8 + t8;
    float v[4], s = 0.f, sq = 0.f;
#pragma unroll
    for (int j = 0; j < 4; ++j) {
      v[j] = xraw[t * C_CH + lane * 4 + j];
      s += v[j]; sq += v[j] * v[j];
    }
    for (int msk = 16; msk >= 1; msk >>= 1) {
      s  += __shfl_xor(s,  msk, 32);
      sq += __shfl_xor(sq, msk, 32);
    }
    float mu   = s * (1.f / 128.f);
    float var  = sq * (1.f / 128.f) - mu * mu;
    float rstd = __frsqrt_rn(var + 1e-5f);
#pragma unroll
    for (int j = 0; j < 4; ++j) {
      int c = lane * 4 + j;
      xn[t * C_CH + c] = (half_t)((v[j] - mu) * rstd * ln_g[c] + ln_b[c]);
    }
  }
  __syncthreads();

  // ---- phase C: QKV GEMM  qkv[64][384] = xn[64][128] @ W_in^T  (96 tiles / 8 waves) ----
  for (int j = wv; j < 96; j += 8) {
    int mt = j / 24, nt = j % 24;
    v8f acc = {};
#pragma unroll
    for (int k = 0; k < 4; ++k) {
      v16h a  = load_A_tile(xn, mt * 16, k * 32, C_CH);
      v16h bb = load_B_tile_T(w_in, nt * 16, k * 32, C_CH);  // W_in[n][k]
      acc = wmma16(a, bb, acc);
    }
    int n = nt * 16 + ln16;
    float bias = b_in[n];
#pragma unroll
    for (int r = 0; r < 8; ++r) {
      int t = mt * 16 + r + 8 * hl;
      qkv[t * 384 + n] = (half_t)(acc[r] + bias);
    }
  }
  __syncthreads();

  // ---- phase D: scores + softmax. 2 waves per head; each wave owns 32 rows ----
  const int head = wv >> 1;
  const int mb   = (wv & 1) * 32;
  const half_t* qp = qkv + head * 32;
  const half_t* kp = qkv + 128 + head * 32;
  const half_t* vp = qkv + 256 + head * 32;

  float sc[2][4][8];
#pragma unroll
  for (int mt = 0; mt < 2; ++mt) {
    v16h a = load_A_tile(qp, mb + mt * 16, 0, 384);   // Q tile, K=hd=32 in one step
#pragma unroll
    for (int nt = 0; nt < 4; ++nt) {
      v16h bb = load_B_tile_T(kp, nt * 16, 0, 384);   // K^T column = contiguous K-row
      v8f acc = {};
      acc = wmma16(a, bb, acc);
#pragma unroll
      for (int r = 0; r < 8; ++r) sc[mt][nt][r] = acc[r] * 0.17677669529f; // 1/sqrt(32)
    }
  }
  // softmax per row: a row lives in one 16-lane N-group; xor masks 1..8 stay in-group
#pragma unroll
  for (int mt = 0; mt < 2; ++mt) {
#pragma unroll
    for (int r = 0; r < 8; ++r) {
      float mx = fmaxf(fmaxf(sc[mt][0][r], sc[mt][1][r]), fmaxf(sc[mt][2][r], sc[mt][3][r]));
      for (int msk = 8; msk >= 1; msk >>= 1) mx = fmaxf(mx, __shfl_xor(mx, msk, 32));
      float sum = 0.f;
#pragma unroll
      for (int nt = 0; nt < 4; ++nt) { sc[mt][nt][r] = __expf(sc[mt][nt][r] - mx); sum += sc[mt][nt][r]; }
      for (int msk = 8; msk >= 1; msk >>= 1) sum += __shfl_xor(sum, msk, 32);
      float inv = __frcp_rn(sum);
      int row = mb + mt * 16 + r + 8 * hl;
#pragma unroll
      for (int nt = 0; nt < 4; ++nt)
        attn[head * 4096 + row * 64 + nt * 16 + ln16] = (half_t)(sc[mt][nt][r] * inv);
    }
  }
  // wave-local RAW on LDS (attn) is kept in order by hardware; no barrier needed.

  // ---- phase E: o = attn @ V  (each wave: its 32 rows x full hd=32) ----
  const half_t* attn_h = attn + head * 4096;
#pragma unroll
  for (int mt = 0; mt < 2; ++mt) {
#pragma unroll
    for (int nt = 0; nt < 2; ++nt) {
      v8f acc = {};
#pragma unroll
      for (int kk = 0; kk < 2; ++kk) {
        v16h a  = load_A_tile(attn_h, mb + mt * 16, kk * 32, 64);
        v16h bb = load_B_tile_KM(vp, kk * 32, nt * 16, 384);   // V[k][d], k-major
        acc = wmma16(a, bb, acc);
      }
      int d = head * 32 + nt * 16 + ln16;
#pragma unroll
      for (int r = 0; r < 8; ++r) {
        int t = mb + mt * 16 + r + 8 * hl;
        osh[t * C_CH + d] = (half_t)acc[r];
      }
    }
  }
  __syncthreads();

  // ---- phase F: out_proj + bias + residual(xraw) -> y_h[b][spatial][c] (f16) ----
  {
    int mt  = wv >> 1;          // 4 token tiles
    int ntb = (wv & 1) * 4;     // 8 channel tiles split 4+4
#pragma unroll
    for (int ntl = 0; ntl < 4; ++ntl) {
      int nt = ntb + ntl;
      v8f acc = {};
#pragma unroll
      for (int k = 0; k < 4; ++k) {
        v16h a  = load_A_tile(osh, mt * 16, k * 32, C_CH);
        v16h bb = load_B_tile_T(w_out, nt * 16, k * 32, C_CH);
        acc = wmma16(a, bb, acc);
      }
      int c = nt * 16 + ln16;
      float bias = b_out[c];
#pragma unroll
      for (int r = 0; r < 8; ++r) {
        int t = mt * 16 + r + 8 * hl;
        float val = acc[r] + bias + xraw[t * C_CH + c];
        int ld2 = t >> 4, lh = (t >> 2) & 3, lw = t & 3;
        size_t gp = (size_t)(wd * 4 + ld2) * HW + (wh * 4 + lh) * 64 + (ww * 4 + lw);
        // channel-contiguous layout: lanes of one row write 16 consecutive halves
        y_h[((size_t)b * DHW + gp) * C_CH + c] = (half_t)val;
      }
    }
  }
}

// ---------- kernel 2: 1x1x1 conv (GEMM over channels) + bias + residual ----------
// y tile fetched by the Tensor Data Mover: 2D tile, 64 rows (spatial) x 128 ch
// (contiguous), delivered to LDS as [p][c] -> directly WMMA-B friendly.
__global__ void __launch_bounds__(256)
conv_residual_kernel(const half_t* __restrict__ y_h, const half_t* __restrict__ w_conv,
                     const float* __restrict__ conv_b, const float* __restrict__ x,
                     float* __restrict__ out) {
  __shared__ half_t y_s[64 * C_CH];   // [p][c]

  const int tid  = threadIdx.x;
  const int lane = tid & 31;
  const int wv   = tid >> 5;
  const int hl   = (lane >> 4) & 1;
  const int ln16 = lane & 15;

  const int bid = blockIdx.x;               // 8192 = 2 batches * 4096 spatial tiles
  const int b   = bid >> 12;
  const int p0  = (bid & 4095) * 64;
  const size_t base = (size_t)b * C_CH * DHW;         // x/out: [B][C][DHW]
  const size_t ybase = ((size_t)b * DHW + p0) * C_CH; // y:     [B][DHW][C]

  if (wv == 0) {
    // ---- Tensor DMA descriptor (CDNA5 ISA) ----
    unsigned long long ga = (unsigned long long)(uintptr_t)(y_h + ybase);
    unsigned int lds_addr = (unsigned int)(uintptr_t)(&y_s[0]);
    u32x4 g0;
    g0[0] = 1u;                                  // count=1, user descriptor
    g0[1] = lds_addr;                            // LDS byte address
    g0[2] = (unsigned int)(ga & 0xFFFFFFFFull);  // global_addr[31:0]
    g0[3] = (unsigned int)((ga >> 32) & 0x1FFFFFFull) | 0x80000000u; // addr[56:32] | type=2
    i32x8 g1;
    g1[0] = 0x00010000;          // data_size=1 (2B), no multicast/pad/iterate
    g1[1] = (int)(128u << 16);   // tensor_dim0[15:0]=128 (row length, elems)
    g1[2] = (int)(64u << 16);    // tensor_dim0[31:16]=0 | tensor_dim1[15:0]=64 rows
    g1[3] = (int)(128u << 16);   // tensor_dim1[31:16]=0 | tile_dim0=128
    g1[4] = 64;                  // tile_dim1=64, tile_dim2=0
    g1[5] = 128;                 // tensor_dim0_stride[31:0]=128 (elems)
    g1[6] = 0;                   // dim0_stride[47:32]=0, dim1_stride[15:0]=0
    g1[7] = 0;
    i32x4 z4 = {};
    i32x8 z8 = {};
    // 6-arg variant (clang-23 / therock-10.0 headers): groups 0..3 + extra group + cpol
    __builtin_amdgcn_tensor_load_to_lds(g0, g1, z4, z4, z8, 0);   // TENSOR_LOAD_TO_LDS
    __builtin_amdgcn_s_wait_tensorcnt((short)0);
  }
  __syncthreads();

  int mt = wv;                                // 8 output-channel tiles, one per wave
#pragma unroll
  for (int nt = 0; nt < 4; ++nt) {
    v8f acc = {};
#pragma unroll
    for (int k = 0; k < 4; ++k) {
      v16h a  = load_A_tile(w_conv, mt * 16, k * 32, C_CH);   // conv_w[o][c]
      v16h bb = load_B_tile_T(y_s, nt * 16, k * 32, C_CH);    // y[p][c]
      acc = wmma16(a, bb, acc);
    }
    int p = p0 + nt * 16 + ln16;
#pragma unroll
    for (int r = 0; r < 8; ++r) {
      int o = mt * 16 + r + 8 * hl;
      size_t idx = base + (size_t)o * DHW + p;
      out[idx] = acc[r] + conv_b[o] + x[idx];
    }
  }
}

// ---------- launcher ----------
extern "C" void kernel_launch(void* const* d_in, const int* in_sizes, int n_in,
                              void* d_out, int out_size, void* d_ws, size_t ws_size,
                              hipStream_t stream) {
  const float* x      = (const float*)d_in[0];
  const float* ln_g   = (const float*)d_in[1];
  const float* ln_b   = (const float*)d_in[2];
  const float* w_in   = (const float*)d_in[3];
  const float* b_in   = (const float*)d_in[4];
  const float* w_out  = (const float*)d_in[5];
  const float* b_out  = (const float*)d_in[6];
  const float* w_conv = (const float*)d_in[7];
  const float* b_conv = (const float*)d_in[8];

  // workspace: f16 weights (49152 + 16384 + 16384 halves) then f16 y tensor [B][DHW][C]
  half_t* w_in_h   = (half_t*)d_ws;
  half_t* w_out_h  = w_in_h + 49152;
  half_t* w_conv_h = w_out_h + 16384;
  half_t* y_h      = w_conv_h + 16384;      // 2*262144*128 halves = 128 MiB

  convert_weights_kernel<<<320, 256, 0, stream>>>(w_in, w_out, w_conv, w_in_h);
  win_attn_kernel<<<8192, 256, 147456, stream>>>(x, ln_g, ln_b, w_in_h, b_in,
                                                 w_out_h, b_out, y_h);
  conv_residual_kernel<<<8192, 256, 0, stream>>>(y_h, w_conv_h, b_conv, x, (float*)d_out);
}
